// OctonionHeadMixerFused_28484223108019
// MI455X (gfx1250) — compile-verified, matrix operand order checked by reading, not run on hardware
//
#include <hip/hip_runtime.h>

// Octonion head mixer: y[b,i] = (Σ_j sign[i,j] · x[b,j] @ W[widx[i,j]]) * beta
// B=8, H=8, T=4096, D=128.
// One WG per (b, i, 512-row T chunk); Wcat[i] staged in 256 KB LDS as bf16
// WMMA B-fragments; each wave computes a 32x128 output tile (2 M-tiles) so
// every LDS B-fragment feeds two v_wmma_f32_16x16x32_bf16.

#define T_DIM 4096
#define D_DIM 128

typedef __attribute__((ext_vector_type(16))) __bf16 v16bf;
typedef __attribute__((ext_vector_type(8)))  float  v8f;

__device__ __constant__ int c_widx[64] = {
  0,1,2,3,4,5,6,7,
  1,0,3,2,5,4,7,6,
  2,3,0,1,6,7,4,5,
  3,2,1,0,7,6,5,4,
  4,5,6,7,0,1,2,3,
  5,4,7,6,1,0,3,2,
  6,7,4,5,2,3,0,1,
  7,6,5,4,3,2,1,0};

__device__ __constant__ float c_sign[64] = {
  +1,-1,-1,-1,-1,-1,-1,-1,
  +1,+1,+1,-1,+1,-1,-1,+1,
  +1,-1,+1,+1,+1,+1,-1,-1,
  +1,+1,-1,+1,+1,-1,+1,-1,
  +1,-1,-1,-1,+1,+1,+1,+1,
  +1,+1,-1,+1,-1,+1,-1,+1,
  +1,+1,+1,-1,-1,+1,+1,-1,
  +1,-1,+1,+1,-1,-1,+1,+1};

// f32 pair -> packed bf16 dword (a in [15:0], b in [31:16]).
// Round-half-up (+0x8000) then grab high halves with one v_perm_b32.
__device__ __forceinline__ unsigned int bfpack(float a, float b) {
  unsigned int ua = __float_as_uint(a) + 0x8000u;
  unsigned int ub = __float_as_uint(b) + 0x8000u;
  return __builtin_amdgcn_perm(ub, ua, 0x07060302u);
}

union Frag16 {
  v16bf h;
  unsigned int u[8];
  uint4 q[2];
};

__global__ __launch_bounds__(512)
void octo_wmma_kernel(const float* __restrict__ x, const float* __restrict__ W,
                      const float* __restrict__ beta, float* __restrict__ y) {
  extern __shared__ unsigned int lds_u32[];   // 65536 dwords = 256 KB

  // Adjacent blocks differ in i -> they share the same x chunk (L2 reuse).
  const int i     = blockIdx.x & 7;           // output head
  const int rest  = blockIdx.x >> 3;
  const int chunk = rest & 7;                 // 512-row T chunk
  const int b     = rest >> 3;                // batch

  // ---- Stage signed/permuted Wcat[i] into LDS as bf16 B-fragments ----
  // Tile tI = (j*4 + kt)*8 + nt is a 32x16 (KxN) tile, 1 KB; logical lane L
  // reads its 8 fragment dwords contiguously at tile_base + L*32 bytes.
  // B-fragment layout (ISA 7.12.2, 16-bit): lane n (0-15) holds col n with
  // K = kt*32 + {0..7, 16..23}; lane n+16 holds K = kt*32 + {8..15, 24..31}.
  for (int idx = threadIdx.x; idx < 65536; idx += 512) {
    const int r  = idx & 7;                   // fragment VGPR index
    const int L  = (idx >> 3) & 31;           // logical lane
    const int tI = idx >> 8;
    const int nt = tI & 7;
    const int kt = (tI >> 3) & 3;
    const int j  = tI >> 5;
    const int wd = c_widx[i * 8 + j];
    const float sg = c_sign[i * 8 + j];
    const int n  = nt * 16 + (L & 15);
    const int kh = L >> 4;
    const int kk = kt * 32 + kh * 8 + ((r < 4) ? 2 * r : 16 + 2 * (r - 4));
    const float* wp = W + ((size_t)wd * D_DIM + kk) * D_DIM + n;
    lds_u32[idx] = bfpack(sg * wp[0], sg * wp[D_DIM]);
  }
  __syncthreads();

  const int wave  = threadIdx.x >> 5;
  const int lane  = threadIdx.x & 31;
  const int row   = lane & 15;                // A-matrix row within M tile
  const int khalf = lane >> 4;                // which K half this lane loads

  const int t0 = chunk * 512 + wave * 32;     // 32-row (2 M-tile) wave tile

  const v8f vzero = {0.f, 0.f, 0.f, 0.f, 0.f, 0.f, 0.f, 0.f};
  v8f acc0[8], acc1[8];
  #pragma unroll
  for (int nt = 0; nt < 8; ++nt) { acc0[nt] = vzero; acc1[nt] = vzero; }

  #pragma unroll 1
  for (int j = 0; j < 8; ++j) {
    // A rows for this lane: x[b, j, t0(+16)+row, khalf*8 + ...]
    const float* xj =
        x + (((size_t)(b * 8 + j)) * T_DIM + (t0 + row)) * D_DIM + khalf * 8;
    if (j < 7) {  // speculative prefetch of next head's rows
      __builtin_prefetch(xj + (size_t)T_DIM * D_DIM, 0, 1);
      __builtin_prefetch(xj + (size_t)T_DIM * D_DIM + 16 * D_DIM, 0, 1);
    }

    #pragma unroll
    for (int kt = 0; kt < 4; ++kt) {
      // 16 fp32 per lane per M-tile: K = kt*32 + khalf*8 + {0..7} and +16
      const float* s0 = xj + kt * 32;
      const float* s1 = s0 + 16 * D_DIM;      // second M-tile, rows +16
      const float4 f0 = ((const float4*)s0)[0];
      const float4 f1 = ((const float4*)s0)[1];
      const float4 f2 = ((const float4*)(s0 + 16))[0];
      const float4 f3 = ((const float4*)(s0 + 16))[1];
      const float4 g0 = ((const float4*)s1)[0];
      const float4 g1 = ((const float4*)s1)[1];
      const float4 g2 = ((const float4*)(s1 + 16))[0];
      const float4 g3 = ((const float4*)(s1 + 16))[1];

      Frag16 a0, a1;
      a0.u[0] = bfpack(f0.x, f0.y);  a0.u[1] = bfpack(f0.z, f0.w);
      a0.u[2] = bfpack(f1.x, f1.y);  a0.u[3] = bfpack(f1.z, f1.w);
      a0.u[4] = bfpack(f2.x, f2.y);  a0.u[5] = bfpack(f2.z, f2.w);
      a0.u[6] = bfpack(f3.x, f3.y);  a0.u[7] = bfpack(f3.z, f3.w);
      a1.u[0] = bfpack(g0.x, g0.y);  a1.u[1] = bfpack(g0.z, g0.w);
      a1.u[2] = bfpack(g1.x, g1.y);  a1.u[3] = bfpack(g1.z, g1.w);
      a1.u[4] = bfpack(g2.x, g2.y);  a1.u[5] = bfpack(g2.z, g2.w);
      a1.u[6] = bfpack(g3.x, g3.y);  a1.u[7] = bfpack(g3.z, g3.w);

      const uint4* bbase =
          (const uint4*)(lds_u32 + (size_t)(j * 4 + kt) * 8 * 256) + lane * 2;

      #pragma unroll
      for (int nt = 0; nt < 8; ++nt) {
        Frag16 bb;
        bb.q[0] = bbase[nt * 64];
        bb.q[1] = bbase[nt * 64 + 1];
        acc0[nt] = __builtin_amdgcn_wmma_f32_16x16x32_bf16(
            false, a0.h, false, bb.h, (short)0, acc0[nt], false, false);
        acc1[nt] = __builtin_amdgcn_wmma_f32_16x16x32_bf16(
            false, a1.h, false, bb.h, (short)0, acc1[nt], false, false);
      }
    }
  }

  // Epilogue: C layout (ISA 7.12.2): VGPR r holds M=r (lanes 0-15, N=lane)
  // and M=r+8 (lanes 16-31, N=lane-16). Scale by beta[col], store fp32.
  float* yrow = y + ((size_t)(b * 8 + i)) * T_DIM * D_DIM;
  #pragma unroll
  for (int nt = 0; nt < 8; ++nt) {
    const int col = nt * 16 + (lane & 15);
    const float bc = beta[col];
    #pragma unroll
    for (int r = 0; r < 8; ++r) {
      const int m  = (lane < 16) ? r : r + 8;
      yrow[(size_t)(t0 + m) * D_DIM + col]      = acc0[nt][r] * bc;
      yrow[(size_t)(t0 + 16 + m) * D_DIM + col] = acc1[nt][r] * bc;
    }
  }
}

extern "C" void kernel_launch(void* const* d_in, const int* in_sizes, int n_in,
                              void* d_out, int out_size, void* d_ws, size_t ws_size,
                              hipStream_t stream) {
  const float* x    = (const float*)d_in[0];   // [8, 8, 4096, 128] fp32
  const float* W    = (const float*)d_in[1];   // [8, 128, 128]     fp32
  const float* beta = (const float*)d_in[2];   // [128]             fp32
  float* y = (float*)d_out;                    // [8, 8, 4096, 128] fp32

  dim3 grid(512);    // 8 i x 8 chunks x 8 b (i fastest -> shared-x adjacency)
  dim3 block(512);   // 16 wave32s, 32 rows each
  octo_wmma_kernel<<<grid, block, 256 * 1024, stream>>>(x, W, beta, y);
}
